// SA_Module_85615878078549
// MI455X (gfx1250) — matrix operand order repaired and checked by reading, not compiled, not run on hardware
//
#include <hip/hip_runtime.h>
#include <hip/hip_bf16.h>

// ---------------------------------------------------------------------------
// Criss-cross (axial) self-attention, bf16 WMMA implementation for gfx1250.
//   B=8, C=512, H=W=128, C2=128
// ---------------------------------------------------------------------------

typedef __attribute__((ext_vector_type(16))) __bf16 v16bf;
typedef __attribute__((ext_vector_type(8)))  __bf16 v8bf;     // 16 B
typedef __attribute__((ext_vector_type(8)))  float  v8f;
typedef __attribute__((ext_vector_type(4)))  unsigned short ushort4v; // 8 B

#define Bc   8
#define Cc   512
#define C2c  128
#define Hc   128
#define Wc   128
#define HWc  16384   // H*W
#define Lc   128     // sequence length along either axis

// fp32 -> bf16 (round to nearest even), raw u16
static __device__ __forceinline__ unsigned short f2bf(float f) {
    unsigned int u = __float_as_uint(f);
    unsigned int r = u + 0x7FFFu + ((u >> 16) & 1u);
    return (unsigned short)(r >> 16);
}

static __device__ __forceinline__ void st4(unsigned short* dst,
                                           unsigned short a, unsigned short b,
                                           unsigned short c, unsigned short d) {
    ushort4v v = {a, b, c, d};
    *reinterpret_cast<ushort4v*>(dst) = v;   // ds_store_b64
}

// ---------------------------------------------------------------------------
// A-fragment: 16x32 bf16 from row-major LDS (ld % 8 == 0 -> 16B aligned).
// Lane half 0: K=0..15; half 1: K=16..31.  Two ds_load_b128 per lane.
// ---------------------------------------------------------------------------
static __device__ __forceinline__ v16bf load_frag_A(const unsigned short* p,
                                                    int row0, int k0, int ld,
                                                    int lane) {
    int hf = (lane >> 4) & 1;
    int r  = lane & 15;
    const unsigned short* base = p + (row0 + r) * ld + k0 + hf * 16;
    v8bf lo = *reinterpret_cast<const v8bf*>(base);        // K + 0..7
    v8bf hi = *reinterpret_cast<const v8bf*>(base + 8);    // K + 8..15
    return __builtin_shufflevector(lo, hi, 0, 1, 2, 3, 4, 5, 6, 7,
                                           8, 9, 10, 11, 12, 13, 14, 15);
}

// ---------------------------------------------------------------------------
// Swizzled B panel: element (k, n) of a Kx128 panel stored so each lane's
// 32x16 fragment is 32 contiguous bytes:
//   idx = (((k>>5)*8 + (n>>4))*32 + (k&16 ? 16 : 0) + (n&15))*16 + (k&15)
// ---------------------------------------------------------------------------
static __device__ __forceinline__ int bsw_idx(int k, int n) {
    int kb = k >> 5;
    int hf = (k >> 4) & 1;
    int i  = k & 15;
    int nt = n >> 4;
    int nl = n & 15;
    return (((kb * 8 + nt) * 32) + hf * 16 + nl) * 16 + i;
}

static __device__ __forceinline__ v16bf load_frag_B_sw(const unsigned short* p,
                                                       int kb, int nt, int lane) {
    const unsigned short* base = p + (((kb * 8 + nt) * 32) + lane) * 16;
    v8bf lo = *reinterpret_cast<const v8bf*>(base);
    v8bf hi = *reinterpret_cast<const v8bf*>(base + 8);
    return __builtin_shufflevector(lo, hi, 0, 1, 2, 3, 4, 5, 6, 7,
                                           8, 9, 10, 11, 12, 13, 14, 15);
}

static __device__ __forceinline__ v8f wmma_bf16(v16bf a, v16bf b, v8f c) {
    return __builtin_amdgcn_wmma_f32_16x16x32_bf16(
        false, a, false, b, (short)0, c, false, false);
}

// 8 N-tiles of one k-step: preload all B fragments, then WMMAs back-to-back.
static __device__ __forceinline__ void mma_row(const unsigned short* sBpanel,
                                               int kb, v16bf af, v8f* acc,
                                               int lane) {
    v16bf bf[8];
#pragma unroll
    for (int nt = 0; nt < 8; ++nt) bf[nt] = load_frag_B_sw(sBpanel, kb, nt, lane);
#pragma unroll
    for (int nt = 0; nt < 8; ++nt) acc[nt] = wmma_bf16(af, bf[nt], acc[nt]);
}

// ---------------------------------------------------------------------------
// Kernel 1: 1x1-conv projection  out_bf16[b, m, hw] = bf16(W[m,:].x[b,:,hw]+bias[m])
// grid: (HW/128, B), block: 256 (8 waves). Each block -> [128 x 128] tile.
// ---------------------------------------------------------------------------
__global__ void proj_kernel(const float* __restrict__ x,
                            const float* __restrict__ wmat,
                            const float* __restrict__ bias,
                            unsigned short* __restrict__ outp) {
    extern __shared__ char blob[];
    unsigned short* sA = (unsigned short*)blob;            // [128][40] row-major W chunk
    unsigned short* sB = sA + 128 * 40;                    // swizzled X chunk (32x128)
    float*       sBias = (float*)(sB + 32 * 128);          // [128]

    const int hw0  = blockIdx.x * 128;
    const int b    = blockIdx.y;
    const int tid  = threadIdx.x;
    const int lane = tid & 31;
    const int wv   = tid >> 5;
    const int m0   = wv * 16;

    if (tid < 128) sBias[tid] = bias[tid];

    v8f acc[8];
#pragma unroll
    for (int nt = 0; nt < 8; ++nt) {
        v8f z = {0.f, 0.f, 0.f, 0.f, 0.f, 0.f, 0.f, 0.f};
        acc[nt] = z;
    }

    for (int kc = 0; kc < Cc; kc += 32) {
        // W chunk: [128 m][32 k], float4 global loads -> b64 LDS stores
        for (int i = tid; i < 128 * 8; i += 256) {
            int m  = i >> 3;
            int k4 = (i & 7) * 4;
            const float4 w4 = *reinterpret_cast<const float4*>(
                wmat + m * Cc + kc + k4);
            st4(sA + m * 40 + k4, f2bf(w4.x), f2bf(w4.y), f2bf(w4.z), f2bf(w4.w));
        }
        // X chunk: [32 k][128 hw], swizzled for B fragments
        for (int i = tid; i < 8 * 128; i += 256) {
            int p  = i & 127;
            int k0 = (i >> 7) * 4;
            unsigned short v[4];
#pragma unroll
            for (int j = 0; j < 4; ++j)
                v[j] = f2bf(x[(long)(b * Cc + kc + k0 + j) * HWc + hw0 + p]);
            st4(sB + bsw_idx(k0, p), v[0], v[1], v[2], v[3]);
        }
        __syncthreads();

        v16bf af = load_frag_A(sA, m0, 0, 40, lane);
        mma_row(sB, 0, af, acc, lane);
        __syncthreads();
    }

    const int hf = lane >> 4;
    const int nl = lane & 15;
    float bv[8];
#pragma unroll
    for (int r = 0; r < 8; ++r) bv[r] = sBias[m0 + r + 8 * hf];

#pragma unroll
    for (int nt = 0; nt < 8; ++nt) {
#pragma unroll
        for (int r = 0; r < 8; ++r) {
            int m = m0 + r + 8 * hf;
            long dst = (long)(b * C2c + m) * HWc + hw0 + nt * 16 + nl;
            outp[dst] = f2bf(acc[nt][r] + bv[r]);
        }
    }
}

// ---------------------------------------------------------------------------
// Kernel 2: axial attention along H (axisH=1) or W (axisH=0).
// grid: (L, B); one block = one (b, slice). block: 256 threads (8 waves).
//   energy = K^T @ Q [128x128], row softmax, out = attn @ X^T.
// Element (c,p) of a slice lives at (b*C'+c)*HW + pos(p):
//   pos(p) = p*W + s  (axis H, slice = column w=s)
//   pos(p) = s*W + p  (axis W, slice = row h=s)
// ---------------------------------------------------------------------------
__global__ void attn_axis_kernel(const float* __restrict__ x,
                                 const unsigned short* __restrict__ qb,
                                 const unsigned short* __restrict__ kb,
                                 float* __restrict__ out,
                                 int axisH) {
    extern __shared__ unsigned short smem[];
    unsigned short* sKT   = smem;                    // [128 p][136] A: K^T, row-major
    unsigned short* sQ    = smem + 128 * 136;        // swizzled B: Q (128k x 128n)
    unsigned short* sAttn = sQ + 128 * 128;          // [128 m][136] A: attn, row-major
    unsigned short* sXT   = sAttn + 128 * 136;       // swizzled B: X^T chunk

    const int s    = blockIdx.x;
    const int b    = blockIdx.y;
    const int tid  = threadIdx.x;
    const int lane = tid & 31;
    const int wv   = tid >> 5;
    const int m0   = wv * 16;
    const int hf   = lane >> 4;
    const int nl   = lane & 15;

    // ---- Phase 1: gather K^T (row-major) and Q (swizzled) slices ----
    for (int i = tid; i < 32 * 128; i += 256) {
        int p  = i & 127;
        int c0 = (i >> 7) * 4;
        unsigned short qv[4], kv[4];
#pragma unroll
        for (int j = 0; j < 4; ++j) {
            long src = (long)(b * C2c + c0 + j) * HWc +
                       (axisH ? p * Wc + s : s * Wc + p);
            qv[j] = qb[src];
            kv[j] = kb[src];
        }
        st4(sKT + p * 136 + c0, kv[0], kv[1], kv[2], kv[3]);
        st4(sQ + bsw_idx(c0, p), qv[0], qv[1], qv[2], qv[3]);
    }
    __syncthreads();

    // ---- Phase 2a: energy[m0..m0+15][:] = K^T @ Q ----
    v8f acc[8];
#pragma unroll
    for (int nt = 0; nt < 8; ++nt) {
        v8f z = {0.f, 0.f, 0.f, 0.f, 0.f, 0.f, 0.f, 0.f};
        acc[nt] = z;
    }
    for (int kbk = 0; kbk < 4; ++kbk) {
        v16bf af = load_frag_A(sKT, m0, kbk * 32, 136, lane);
        mma_row(sQ, kbk, af, acc, lane);
    }
    __syncthreads();

    // ---- Phase 2b: row softmax in registers (row m = m0 + r + 8*hf) ----
    float mx[8];
#pragma unroll
    for (int r = 0; r < 8; ++r) mx[r] = acc[0][r];
#pragma unroll
    for (int nt = 1; nt < 8; ++nt)
#pragma unroll
        for (int r = 0; r < 8; ++r) mx[r] = fmaxf(mx[r], acc[nt][r]);
    for (int msk = 1; msk <= 8; msk <<= 1)
#pragma unroll
        for (int r = 0; r < 8; ++r)
            mx[r] = fmaxf(mx[r], __shfl_xor(mx[r], msk, 32));

    float sm[8];
#pragma unroll
    for (int r = 0; r < 8; ++r) sm[r] = 0.f;
#pragma unroll
    for (int nt = 0; nt < 8; ++nt)
#pragma unroll
        for (int r = 0; r < 8; ++r) {
            acc[nt][r] = __expf(acc[nt][r] - mx[r]);
            sm[r] += acc[nt][r];
        }
    for (int msk = 1; msk <= 8; msk <<= 1)
#pragma unroll
        for (int r = 0; r < 8; ++r) sm[r] += __shfl_xor(sm[r], msk, 32);

#pragma unroll
    for (int r = 0; r < 8; ++r) {
        float inv = 1.0f / sm[r];
        int m = m0 + r + 8 * hf;
#pragma unroll
        for (int nt = 0; nt < 8; ++nt)
            sAttn[m * 136 + nt * 16 + nl] = f2bf(acc[nt][r] * inv);
    }
    __syncthreads();

    // ---- Phase 3: O[m, c] = attn @ X^T, over C in 4 chunks of 128 ----
    for (int cc = 0; cc < Cc; cc += 128) {
        for (int i = tid; i < 32 * 128; i += 256) {
            int cl = i >> 5;
            int g0 = (i & 31) * 4;
            unsigned short v[4];
#pragma unroll
            for (int j = 0; j < 4; ++j) {
                long src = (long)(b * Cc + cc + cl) * HWc +
                           (axisH ? (g0 + j) * Wc + s : s * Wc + g0 + j);
                v[j] = f2bf(x[src]);
            }
            st4(sXT + bsw_idx(g0, cl), v[0], v[1], v[2], v[3]);
        }
        __syncthreads();

        v8f oacc[8];
#pragma unroll
        for (int nt = 0; nt < 8; ++nt) {
            v8f z = {0.f, 0.f, 0.f, 0.f, 0.f, 0.f, 0.f, 0.f};
            oacc[nt] = z;
        }
        for (int gbk = 0; gbk < 4; ++gbk) {
            v16bf af = load_frag_A(sAttn, m0, gbk * 32, 136, lane);
            mma_row(sXT, gbk, af, oacc, lane);
        }

#pragma unroll
        for (int nt = 0; nt < 8; ++nt) {
#pragma unroll
            for (int r = 0; r < 8; ++r) {
                int m = m0 + r + 8 * hf;              // position along the axis
                int c = cc + nt * 16 + nl;            // channel
                long dst = (long)(b * Cc + c) * HWc +
                           (axisH ? m * Wc + s : s * Wc + m);
                if (axisH) out[dst] = oacc[nt][r];
                else       atomicAdd(&out[dst], oacc[nt][r]); // non-returning f32 atomic
            }
        }
        __syncthreads();
    }
}

// ---------------------------------------------------------------------------
extern "C" void kernel_launch(void* const* d_in, const int* in_sizes, int n_in,
                              void* d_out, int out_size, void* d_ws, size_t ws_size,
                              hipStream_t stream) {
    const float* x  = (const float*)d_in[0];
    const float* wq = (const float*)d_in[1];
    const float* bq = (const float*)d_in[2];
    const float* wk = (const float*)d_in[3];
    const float* bk = (const float*)d_in[4];
    float* out = (float*)d_out;

    unsigned short* qb = (unsigned short*)d_ws;              // [B,C2,H,W] bf16
    unsigned short* kb = qb + (size_t)Bc * C2c * HWc;        // [B,C2,H,W] bf16

    dim3 blk(256);

    // projections -> bf16 q / k workspaces
    dim3 gp(HWc / 128, Bc);
    size_t smem_proj = (size_t)128 * 40 * 2 + (size_t)32 * 128 * 2 + 128 * 4;
    proj_kernel<<<gp, blk, smem_proj, stream>>>(x, wq, bq, qb);
    proj_kernel<<<gp, blk, smem_proj, stream>>>(x, wk, bk, kb);

    // axial attention: H axis writes, W axis accumulates
    dim3 ga(Lc, Bc);
    size_t smem_attn =
        (size_t)(128 * 136 + 128 * 128 + 128 * 136 + 128 * 128) * 2; // 135168 B
    attn_axis_kernel<<<ga, blk, smem_attn, stream>>>(x, qb, kb, out, 1);
    attn_axis_kernel<<<ga, blk, smem_attn, stream>>>(x, qb, kb, out, 0);
}
